// S4A_NS_73418170958498
// MI455X (gfx1250) — matrix operand order repaired and checked by reading, not compiled, not run on hardware
//
#include <hip/hip_runtime.h>
#include <hip/hip_bf16.h>

// ---------------------------------------------------------------------------
// Mamba-style block for MI455X (gfx1250, wave32, WMMA).
//   out = ((mamba(x @ Wd + bd)) @ Wu) + bu
// Weights pre-converted once to bf16, transposed [N][K] (zero-padded to 64).
// GEMMs: v_wmma_f32_16x16x32_bf16, 32x64 per wave, 8 WMMA / wave / K-step,
// B fragments straight from L2 (weights are L2-resident), no LDS/barriers.
// Scan: 1024 sequential steps, parallel over 16*768 channels, B/C via LDS.
// ---------------------------------------------------------------------------

typedef __attribute__((ext_vector_type(16))) __bf16 v16bf;
typedef __attribute__((ext_vector_type(8)))  float  v8f;

__device__ __forceinline__ __bf16 f2bf(float f) {
    union { float f; unsigned u; } v; v.f = f;
    unsigned r = v.u + 0x7FFFu + ((v.u >> 16) & 1u);   // round-to-nearest-even
    union { unsigned short s; __bf16 b; } o;
    o.s = (unsigned short)(r >> 16);
    return o.b;
}

// A fragment: contiguous k order used here (B loads match) — lane holds row (lane&15); lanes 0-15:
// K = {0..15}, lanes 16-31: K = {16..31}  (c0 = hi*16).
__device__ __forceinline__ v16bf load_a_frag(const float* __restrict__ abase, int c0) {
    const float4* ap0 = (const float4*)(abase + c0);
    const float4* ap1 = (const float4*)(abase + c0 + 8);
    float4 a00 = ap0[0], a01 = ap0[1];
    float4 a10 = ap1[0], a11 = ap1[1];
    v16bf af;
    af[0]  = f2bf(a00.x); af[1]  = f2bf(a00.y); af[2]  = f2bf(a00.z); af[3]  = f2bf(a00.w);
    af[4]  = f2bf(a01.x); af[5]  = f2bf(a01.y); af[6]  = f2bf(a01.z); af[7]  = f2bf(a01.w);
    af[8]  = f2bf(a10.x); af[9]  = f2bf(a10.y); af[10] = f2bf(a10.z); af[11] = f2bf(a10.w);
    af[12] = f2bf(a11.x); af[13] = f2bf(a11.y); af[14] = f2bf(a11.z); af[15] = f2bf(a11.w);
    return af;
}

// ---------------------------------------------------------------------------
// Weight prep: W[K][N] f32 -> Wt[Npad][K] bf16 (cols >= N zero-padded).
// ---------------------------------------------------------------------------
__global__ __launch_bounds__(256)
void wt_convert(const float* __restrict__ W, __bf16* __restrict__ Wt,
                int N, int K)
{
    int id = blockIdx.x * 256 + threadIdx.x;   // over Npad*K
    int c = id / K, k = id - c * K;
    int cs = (c < N) ? c : (N - 1);
    float v = W[k * N + cs];
    v = (c < N) ? v : 0.0f;
    Wt[c * K + k] = f2bf(v);
}

// ---------------------------------------------------------------------------
// C[M,N] = A[M,K] @ Bt^T (+ bias[N]);  Bt is [Npad][K] bf16, Npad % 64 == 0.
// Block: 256 threads = 8 waves; wave tile 32 rows x 64 cols (8 WMMA/K-step).
// grid.x = Npad/64, grid.y = M/256. Requires K % 32 == 0, M % 256 == 0.
// ---------------------------------------------------------------------------
__global__ __launch_bounds__(256)
void gemm_wmma_bf16t(const float* __restrict__ A, const __bf16* __restrict__ Bt,
                     const float* __restrict__ bias, float* __restrict__ C,
                     int N, int K)
{
    const int tid  = threadIdx.x;
    const int wave = tid >> 5;
    const int lane = tid & 31;
    const int l15  = lane & 15;
    const int hi   = lane >> 4;            // 0: lanes 0-15, 1: lanes 16-31
    const int bn   = blockIdx.x * 64;
    const int bm   = blockIdx.y * 256 + wave * 32;
    const int c0   = hi * 16;              // A-frag K sub-offset
    const int kk   = hi * 16;              // B-frag K sub-offset
    const bool fullN = (bn + 64 <= N);     // uniform (false only for N=56)

    const float* a0 = A + (bm + l15) * K;        // m-tile 0 row
    const float* a1 = a0 + 16 * K;               // m-tile 1 row
    const __bf16* bp0 = Bt + (bn + l15     ) * K + kk;
    const __bf16* bp1 = Bt + (bn + l15 + 16) * K + kk;
    const __bf16* bp2 = Bt + (bn + l15 + 32) * K + kk;
    const __bf16* bp3 = Bt + (bn + l15 + 48) * K + kk;

    v8f acc[2][4] = {};

    for (int k0 = 0; k0 < K; k0 += 32) {
        if (k0 + 32 < K) {
            __builtin_prefetch(a0 + k0 + 32, 0, 1);
            __builtin_prefetch(a1 + k0 + 32, 0, 1);
        }
        v16bf bf[4];
        bf[0] = *(const v16bf*)(bp0 + k0);
        bf[1] = *(const v16bf*)(bp1 + k0);
        bf[2] = *(const v16bf*)(bp2 + k0);
        bf[3] = *(const v16bf*)(bp3 + k0);

        v16bf af0 = load_a_frag(a0 + k0, c0);
        v16bf af1 = load_a_frag(a1 + k0, c0);

        #pragma unroll
        for (int nt = 0; nt < 4; ++nt) {
            acc[0][nt] = __builtin_amdgcn_wmma_f32_16x16x32_bf16(
                false, af0, false, bf[nt], (short)0, acc[0][nt], false, false);
            acc[1][nt] = __builtin_amdgcn_wmma_f32_16x16x32_bf16(
                false, af1, false, bf[nt], (short)0, acc[1][nt], false, false);
        }
    }

    // epilogue: D layout — VGPR r -> row (hi*8 + r), col = lane&15
    #pragma unroll
    for (int mt = 0; mt < 2; ++mt) {
        const int rbase = bm + mt * 16 + hi * 8;
        if (fullN) {
            #pragma unroll
            for (int nt = 0; nt < 4; ++nt) {
                int col = bn + nt * 16 + l15;
                float bv = bias ? bias[col] : 0.0f;
                #pragma unroll
                for (int r8 = 0; r8 < 8; ++r8)
                    C[(rbase + r8) * N + col] = acc[mt][nt][r8] + bv;
            }
        } else {
            #pragma unroll
            for (int nt = 0; nt < 4; ++nt) {
                int col = bn + nt * 16 + l15;
                if (col < N) {
                    float bv = bias ? bias[col] : 0.0f;
                    #pragma unroll
                    for (int r8 = 0; r8 < 8; ++r8)
                        C[(rbase + r8) * N + col] = acc[mt][nt][r8] + bv;
                }
            }
        }
    }
}

// ---------------------------------------------------------------------------
// Depthwise causal conv (K=4) + SiLU:  u[m,e] from xm = xz[:, :768]
// grid = (16384, 3), block = 256: token m is block-uniform -> scalar guard.
// ---------------------------------------------------------------------------
__global__ __launch_bounds__(256)
void conv_silu_kernel(const float* __restrict__ xz, const float* __restrict__ conv_w,
                      const float* __restrict__ conv_b, float* __restrict__ u)
{
    const int m = blockIdx.x;                          // token index, uniform
    const int e = blockIdx.y * 256 + threadIdx.x;      // channel
    const int t = m & 1023;
    float acc = conv_b[e];
    if (t >= 3) {                                      // uniform branch
        #pragma unroll
        for (int j = 0; j < 4; ++j)
            acc += xz[(m - 3 + j) * 1536 + e] * conv_w[e * 4 + j];
    } else {
        #pragma unroll
        for (int j = 0; j < 4; ++j)
            if (t - 3 + j >= 0)
                acc += xz[(m - 3 + j) * 1536 + e] * conv_w[e * 4 + j];
    }
    u[m * 768 + e] = acc / (1.0f + __expf(-acc));      // silu
}

// ---------------------------------------------------------------------------
// dt = softplus(xdbl[:, :24] @ Wdt + bdt)    (K=24: VALU dot, scalar A reads)
// grid = (16384, 3), block = 256.
// ---------------------------------------------------------------------------
__global__ __launch_bounds__(256)
void dt_kernel(const float* __restrict__ xdbl, const float* __restrict__ Wdt,
               const float* __restrict__ bdt, float* __restrict__ dt)
{
    const int m = blockIdx.x;                          // token index, uniform
    const int e = blockIdx.y * 256 + threadIdx.x;      // channel
    float acc = bdt[e];
    const float* xr = xdbl + m * 56;                   // uniform -> scalar loads
    #pragma unroll
    for (int r = 0; r < 24; ++r)
        acc += xr[r] * Wdt[r * 768 + e];
    dt[m * 768 + e] = (acc > 20.0f) ? acc : __logf(1.0f + __expf(acc));
}

// ---------------------------------------------------------------------------
// Selective scan: grid (3, 16), block 256 -> one thread per (b, e) channel.
// h-state (N=16) in registers; B/C (shared across E) staged via LDS in
// 8-step chunks. Fuses  y = (scan + u*D) * silu(z).
// ---------------------------------------------------------------------------
__global__ __launch_bounds__(256)
void scan_kernel(const float* __restrict__ xdbl, const float* __restrict__ dt,
                 const float* __restrict__ u, const float* __restrict__ xz,
                 const float* __restrict__ A_log, const float* __restrict__ Dp,
                 float* __restrict__ y)
{
    const int b = blockIdx.y;
    const int e = blockIdx.x * 256 + threadIdx.x;      // < 768

    __shared__ float sB[8][16];
    __shared__ float sC[8][16];

    float hreg[16];
    float Ae[16];
    #pragma unroll
    for (int n = 0; n < 16; ++n) {
        hreg[n] = 0.0f;
        Ae[n] = -__expf(A_log[e * 16 + n]);
    }
    const float De = Dp[e];

    for (int t0 = 0; t0 < 1024; t0 += 8) {
        // stage 8 timesteps of B (16) and C (16) for this batch
        {
            int i  = threadIdx.x;                      // 256 threads, 256 floats
            int tt = i >> 5, j = i & 31;
            int base = (b * 1024 + t0 + tt) * 56;
            if (j < 16) sB[tt][j]      = xdbl[base + 24 + j];
            else        sC[tt][j - 16] = xdbl[base + 40 + (j - 16)];
        }
        __syncthreads();

        #pragma unroll 1
        for (int tt = 0; tt < 8; ++tt) {
            int   mi  = b * 1024 + t0 + tt;
            float dtv = dt[mi * 768 + e];
            float uv  = u [mi * 768 + e];
            float zv  = xz[mi * 1536 + 768 + e];
            float dtu = dtv * uv;
            float yv  = 0.0f;
            #pragma unroll
            for (int n = 0; n < 16; ++n) {
                float dA = __expf(dtv * Ae[n]);
                hreg[n] = hreg[n] * dA + dtu * sB[tt][n];
                yv += hreg[n] * sC[tt][n];
            }
            yv = yv + uv * De;
            yv = yv * (zv / (1.0f + __expf(-zv)));     // * silu(z)
            y[mi * 768 + e] = yv;
        }
        __syncthreads();
    }
}

// ---------------------------------------------------------------------------
// Launch
// ---------------------------------------------------------------------------
extern "C" void kernel_launch(void* const* d_in, const int* in_sizes, int n_in,
                              void* d_out, int out_size, void* d_ws, size_t ws_size,
                              hipStream_t stream)
{
    const float* x      = (const float*)d_in[0];
    const float* Wd     = (const float*)d_in[1];
    const float* bd     = (const float*)d_in[2];
    const float* Win    = (const float*)d_in[3];
    const float* conv_w = (const float*)d_in[4];
    const float* conv_b = (const float*)d_in[5];
    const float* Wx     = (const float*)d_in[6];
    const float* Wdt    = (const float*)d_in[7];
    const float* bdt    = (const float*)d_in[8];
    const float* A_log  = (const float*)d_in[9];
    const float* Dp     = (const float*)d_in[10];
    const float* Wout   = (const float*)d_in[11];
    const float* Wu     = (const float*)d_in[12];
    const float* bu     = (const float*)d_in[13];
    float* out = (float*)d_out;

    const int M = 16 * 1024;                       // 16384 tokens

    // workspace layout (bytes, all 64B-aligned); dead regions are reused
    char*  ws   = (char*)d_ws;
    float* xz   = (float*)(ws);                    // 16384*1536 f32 = 100.7 MB
    float* u    = (float*)(ws + 100663296);        // 16384* 768 f32 =  50.3 MB
    float* dtb  = (float*)(ws + 150994944);        // 16384* 768 f32 =  50.3 MB
    float* ybuf = (float*)(ws + 201326592);        // 16384* 768 f32 =  50.3 MB
    float* xdbl = (float*)(ws + 251658240);        // 16384*  56 f32 =   3.7 MB
    float* h    = ybuf;                            // 16384*384 — dead before y written
    float* ym   = dtb;                             // reuse dt region after scan

    // bf16 transposed weights [Npad][K]
    __bf16* wt_d   = (__bf16*)(ws + 255328256);    // 384 x768 -> 589824 B
    __bf16* wt_in  = (__bf16*)(ws + 255918080);    // 1536x384 -> 1179648 B
    __bf16* wt_x   = (__bf16*)(ws + 257097728);    //  64 x768 ->  98304 B
    __bf16* wt_out = (__bf16*)(ws + 257196032);    // 384 x768 -> 589824 B
    __bf16* wt_u   = (__bf16*)(ws + 257785856);    // 768 x384 -> 589824 B

    // 0) weight prep (f32 [K][N] -> bf16 [Npad][K])
    wt_convert<<<(384  * 768) / 256, 256, 0, stream>>>(Wd,   wt_d,   384,  768);
    wt_convert<<<(1536 * 384) / 256, 256, 0, stream>>>(Win,  wt_in,  1536, 384);
    wt_convert<<<(64   * 768) / 256, 256, 0, stream>>>(Wx,   wt_x,   56,   768);
    wt_convert<<<(384  * 768) / 256, 256, 0, stream>>>(Wout, wt_out, 384,  768);
    wt_convert<<<(768  * 384) / 256, 256, 0, stream>>>(Wu,   wt_u,   768,  384);

    dim3 blk(256);
    const int GY = M / 256;                        // 64

    // 1) h = x @ Wd + bd                      (16384x768 @ 768x384)
    gemm_wmma_bf16t<<<dim3(384 / 64, GY), blk, 0, stream>>>(x, wt_d, bd, h, 384, 768);

    // 2) xz = h @ Win                         (16384x384 @ 384x1536)
    gemm_wmma_bf16t<<<dim3(1536 / 64, GY), blk, 0, stream>>>(h, wt_in, nullptr, xz, 1536, 384);

    // 3) u = silu(causal_dwconv(xm) + conv_b)
    conv_silu_kernel<<<dim3(M, 3), 256, 0, stream>>>(xz, conv_w, conv_b, u);

    // 4) xdbl = u @ Wx                        (16384x768 @ 768x56, Npad=64)
    gemm_wmma_bf16t<<<dim3(1, GY), blk, 0, stream>>>(u, wt_x, nullptr, xdbl, 56, 768);

    // 5) dt = softplus(xdbl[:, :24] @ Wdt + bdt)
    dt_kernel<<<dim3(M, 3), 256, 0, stream>>>(xdbl, Wdt, bdt, dtb);

    // 6) selective scan (+ u*D, * silu(z)) -> y
    scan_kernel<<<dim3(3, 16), 256, 0, stream>>>(xdbl, dtb, u, xz, A_log, Dp, ybuf);

    // 7) ym = y @ Wout                        (16384x768 @ 768x384)
    gemm_wmma_bf16t<<<dim3(384 / 64, GY), blk, 0, stream>>>(ybuf, wt_out, nullptr, ym, 384, 768);

    // 8) out = ym @ Wu + bu                   (16384x384 @ 384x768)
    gemm_wmma_bf16t<<<dim3(768 / 64, GY), blk, 0, stream>>>(ym, wt_u, bu, out, 768, 384);
}
